// MultiScaleDecoder_deterministic_11373073400245
// MI455X (gfx1250) — matrix operand order repaired and checked by reading, not compiled, not run on hardware
//
#include <hip/hip_runtime.h>
#include <hip/hip_bf16.h>

typedef __attribute__((ext_vector_type(16))) __bf16 v16bf;
typedef __attribute__((ext_vector_type(8)))  float  v8f;

#define B_SZ   256
#define D_SZ   128
#define T_SZ   512
#define H_SZ   512
#define N3H    1536              // 3*H
#define NT_G   (N3H / 16)        // 96 n-tiles for the gate matmuls
#define KB_IH  (D_SZ / 32)       // 4  k-blocks for gi
#define KB_HH  (H_SZ / 32)       // 16 k-blocks for gh
#define NT_FC  (D_SZ / 16)       // 8  n-tiles for fc
#define KB_FC  (H_SZ / 32)       // 16 k-blocks for fc
#define FRAG_USH 512             // 32 lanes * 16 ushorts per B-fragment

#define N_FRAG_IH (KB_IH * NT_G)                       // 384
#define N_FRAG_HH (KB_HH * NT_G)                       // 1536
#define N_FRAG_FC (KB_FC * NT_FC)                      // 128
#define OFF_IH 0
#define OFF_HH (OFF_IH + N_FRAG_IH * FRAG_USH)
#define OFF_FC (OFF_HH + N_FRAG_HH * FRAG_USH)
#define N_FRAG_TOT (N_FRAG_IH + N_FRAG_HH + N_FRAG_FC) // 2048

// ---------- helpers ----------

__device__ __forceinline__ unsigned short f2bf(float f) {
    unsigned int u = __float_as_uint(f);
    unsigned int r = u + 0x7FFFu + ((u >> 16) & 1u);   // round-to-nearest-even
    return (unsigned short)(r >> 16);
}
__device__ __forceinline__ float bf2f(unsigned short h) {
    return __uint_as_float(((unsigned int)h) << 16);
}
__device__ __forceinline__ float fsigm(float x) {
    return 1.0f / (1.0f + __expf(-x));
}
__device__ __forceinline__ float ftanh(float x) {
    float e = __expf(-2.0f * x);
    return (1.0f - e) / (1.0f + e);
}

union BFV {
    v16bf v;
    unsigned int w[8];
    uint4 q[2];
};

// B-fragment: pre-swizzled in workspace; each lane loads 32 contiguous bytes.
__device__ __forceinline__ v16bf load_bfrag(const unsigned short* frag_base, int lane) {
    const uint4* p = reinterpret_cast<const uint4*>(frag_base + lane * 16);
    BFV u;
    u.q[0] = p[0];
    u.q[1] = p[1];
    return u.v;
}

// A-fragment from bf16 rows in LDS. Lane holds row M = lane&15.
// K layout (16-bit A 16x32): lanes<16 hold K {0..7,16..23}, lanes>=16 hold K {8..15,24..31}.
__device__ __forceinline__ v16bf a_from_bf16(const unsigned short* base, int ld,
                                             int kb, int lane) {
    BFV u;
    const int M   = lane & 15;
    const int khi = (lane >= 16) ? 8 : 0;
    const unsigned short* row = base + M * ld;
#pragma unroll
    for (int v = 0; v < 8; ++v) {
        int Kp = kb * 32 + 2 * v + ((v >= 4) ? 8 : 0) + khi;   // even -> 4B aligned
        u.w[v] = *reinterpret_cast<const unsigned int*>(row + Kp);
    }
    return u.v;
}

__device__ __forceinline__ v8f wmma_bf16(v16bf a, v16bf b, v8f c) {
    return __builtin_amdgcn_wmma_f32_16x16x32_bf16(
        /*neg_a=*/false, a, /*neg_b=*/false, b,
        /*c_mod=*/(short)0, c, /*reuse_a=*/false, /*reuse_b=*/false);
}

// ---------- kernel 1: pack fp32 weights into bf16 WMMA B-fragments in ws ----------
// B layout (16-bit 32x16): lane l holds column N = l&15; lanes<16 hold K rows
// kb*32 + {0..7,16..23}, lanes>=16 hold K rows kb*32 + {8..15,24..31}; 16 halves per lane.
__global__ __launch_bounds__(32)
void pack_weights_kernel(const float* __restrict__ W_ih,
                         const float* __restrict__ W_hh,
                         const float* __restrict__ fc_W,
                         unsigned short* __restrict__ ws) {
    const int f    = blockIdx.x;
    const int lane = threadIdx.x;
    const int koff = (lane >= 16) ? 8 : 0;
    const int nsub = lane & 15;

    if (f < N_FRAG_IH) {
        const int kb = f / NT_G, nt = f % NT_G;
        const int n  = nt * 16 + nsub;
        unsigned short* dst = ws + OFF_IH + (size_t)f * FRAG_USH + lane * 16;
#pragma unroll
        for (int j = 0; j < 16; ++j) {
            int k = kb * 32 + koff + j + ((j >= 8) ? 8 : 0);
            dst[j] = f2bf(W_ih[(size_t)n * D_SZ + k]);   // B(k,n) = W_ih[n,k]
        }
    } else if (f < N_FRAG_IH + N_FRAG_HH) {
        const int f2 = f - N_FRAG_IH;
        const int kb = f2 / NT_G, nt = f2 % NT_G;
        const int n  = nt * 16 + nsub;
        unsigned short* dst = ws + OFF_HH + (size_t)f2 * FRAG_USH + lane * 16;
#pragma unroll
        for (int j = 0; j < 16; ++j) {
            int k = kb * 32 + koff + j + ((j >= 8) ? 8 : 0);
            dst[j] = f2bf(W_hh[(size_t)n * H_SZ + k]);   // B(k,n) = W_hh[n,k]
        }
    } else {
        const int f3 = f - N_FRAG_IH - N_FRAG_HH;
        const int kb = f3 / NT_FC, nt = f3 % NT_FC;
        const int n  = nt * 16 + nsub;
        unsigned short* dst = ws + OFF_FC + (size_t)f3 * FRAG_USH + lane * 16;
#pragma unroll
        for (int j = 0; j < 16; ++j) {
            int k = kb * 32 + koff + j + ((j >= 8) ? 8 : 0);
            dst[j] = f2bf(fc_W[(size_t)n * H_SZ + k]);   // B(k,n) = fc_W[n,k]
        }
    }
}

// ---------- kernel 2: persistent batch-tiled GRU rollout ----------
// 16 blocks (one per 16-row batch tile) x 512 threads (16 waves).
// Waves split the 3H gate columns as 32 H-tiles (2 per wave); fc uses waves 0..7.
// Phase A is pure ds_load_b128 (A frags) + global_load_b128 (B frags) + WMMA;
// all fp32<->bf16 conversion happens once per element in phase B / h-update.
__global__ __launch_bounds__(512)
void gru_rollout_kernel(const float* __restrict__ z_in,
                        const float* __restrict__ b_ih,
                        const float* __restrict__ b_hh,
                        const float* __restrict__ fc_b,
                        const unsigned short* __restrict__ wfrag,
                        float* __restrict__ out) {
    __shared__ unsigned short hbuf[2][16 * H_SZ];  // h state, bf16, double-buffered (32 KB)
    __shared__ float          zbuf[16 * D_SZ];     // z state, fp32 master copy (8 KB)
    __shared__ unsigned short zbf [16 * D_SZ];     // z state, bf16 shadow for WMMA A (4 KB)

    const int tid  = threadIdx.x;
    const int lane = tid & 31;
    const int wid  = tid >> 5;
    const int bg0  = blockIdx.x * 16;
    const int nsub = lane & 15;
    const int mhi  = (lane >= 16) ? 8 : 0;

    // ---- init: h = 0, z = z[:, :, 0]; emit t=0 output ----
    for (int i = tid; i < 16 * H_SZ; i += 512) hbuf[0][i] = 0;
    for (int i = tid; i < 16 * D_SZ; i += 512) {
        int M = i / D_SZ, d = i % D_SZ;
        size_t gi = (size_t)(bg0 + M) * D_SZ * T_SZ + (size_t)d * T_SZ;
        float v = z_in[gi];
        zbuf[i] = v;
        zbf[i]  = f2bf(v);
        out[gi] = v;
    }
    __syncthreads();

    // ---- per-wave constants: bias registers for both owned H-tiles + fc bias ----
    float bias_r[2], bias_z[2], bias_in[2], bias_hn[2];
#pragma unroll
    for (int q = 0; q < 2; ++q) {
        int c = (2 * wid + q) * 16 + nsub;
        bias_r[q]  = b_ih[c] + b_hh[c];
        bias_z[q]  = b_ih[H_SZ + c] + b_hh[H_SZ + c];
        bias_in[q] = b_ih[2 * H_SZ + c];
        bias_hn[q] = b_hh[2 * H_SZ + c];
    }
    const float fb = (wid < 8) ? fc_b[wid * 16 + nsub] : 0.0f;

    int cur = 0;
    for (int t = 1; t < T_SZ; ++t) {
        const int nxt = cur ^ 1;

        // ---- phase A: gates + h update (reads zbf, h[cur]; writes h[nxt]) ----
#pragma unroll 1
        for (int q = 0; q < 2; ++q) {
            const int ht = 2 * wid + q;        // H-tile index, columns [ht*16, ht*16+16)
            v8f accr, accz, accin, acchn;
#pragma unroll
            for (int i = 0; i < 8; ++i) {
                accr[i]  = bias_r[q];
                accz[i]  = bias_z[q];
                accin[i] = bias_in[q];
                acchn[i] = bias_hn[q];
            }
            // gi contribution: A = z_prev tile (K = D = 128), bf16 straight from LDS
            for (int kb = 0; kb < KB_IH; ++kb) {
                v16bf a  = a_from_bf16(zbf, D_SZ, kb, lane);
                v16bf br = load_bfrag(wfrag + OFF_IH + (size_t)(kb * NT_G + ht)      * FRAG_USH, lane);
                v16bf bz = load_bfrag(wfrag + OFF_IH + (size_t)(kb * NT_G + 32 + ht) * FRAG_USH, lane);
                v16bf bn = load_bfrag(wfrag + OFF_IH + (size_t)(kb * NT_G + 64 + ht) * FRAG_USH, lane);
                accr  = wmma_bf16(a, br, accr);
                accz  = wmma_bf16(a, bz, accz);
                accin = wmma_bf16(a, bn, accin);
            }
            // gh contribution: A = h tile (K = H = 512)
            for (int kb = 0; kb < KB_HH; ++kb) {
                v16bf a  = a_from_bf16(&hbuf[cur][0], H_SZ, kb, lane);
                v16bf br = load_bfrag(wfrag + OFF_HH + (size_t)(kb * NT_G + ht)      * FRAG_USH, lane);
                v16bf bz = load_bfrag(wfrag + OFF_HH + (size_t)(kb * NT_G + 32 + ht) * FRAG_USH, lane);
                v16bf bn = load_bfrag(wfrag + OFF_HH + (size_t)(kb * NT_G + 64 + ht) * FRAG_USH, lane);
                accr  = wmma_bf16(a, br, accr);
                accz  = wmma_bf16(a, bz, accz);
                acchn = wmma_bf16(a, bn, acchn);
            }
            // gate nonlinearities + convex h update
#pragma unroll
            for (int i = 0; i < 8; ++i) {
                float r  = fsigm(accr[i]);
                float zg = fsigm(accz[i]);
                float nn = ftanh(accin[i] + r * acchn[i]);
                int   M  = i + mhi;
                int   col = ht * 16 + nsub;
                float hold = bf2f(hbuf[cur][M * H_SZ + col]);
                float hnew = (1.0f - zg) * nn + zg * hold;
                hbuf[nxt][M * H_SZ + col] = f2bf(hnew);
            }
        }
        __syncthreads();   // h[nxt] complete; all reads of zbf and h[cur] done

        // ---- phase B: delta = h_new @ fc_W^T + fc_b; z += delta; emit output ----
        if (wid < 8) {
            v8f acc;
#pragma unroll
            for (int i = 0; i < 8; ++i) acc[i] = fb;
            for (int kb = 0; kb < KB_FC; ++kb) {
                v16bf a = a_from_bf16(&hbuf[nxt][0], H_SZ, kb, lane);
                v16bf b = load_bfrag(wfrag + OFF_FC + (size_t)(kb * NT_FC + wid) * FRAG_USH, lane);
                acc = wmma_bf16(a, b, acc);
            }
#pragma unroll
            for (int i = 0; i < 8; ++i) {
                int   M = i + mhi;
                int   d = wid * 16 + nsub;
                float znew = zbuf[M * D_SZ + d] + acc[i];
                zbuf[M * D_SZ + d] = znew;
                zbf [M * D_SZ + d] = f2bf(znew);   // bf16 shadow for next step's gi A-frags
                out[(size_t)(bg0 + M) * D_SZ * T_SZ + (size_t)d * T_SZ + t] = znew;
            }
        }
        __syncthreads();   // z/zbf updated before next step's reads

        cur ^= 1;
    }
}

// ---------- launch ----------
extern "C" void kernel_launch(void* const* d_in, const int* in_sizes, int n_in,
                              void* d_out, int out_size, void* d_ws, size_t ws_size,
                              hipStream_t stream) {
    (void)in_sizes; (void)n_in; (void)out_size; (void)ws_size;
    const float* z    = (const float*)d_in[0];
    // d_in[1] = e : unused by the reference
    const float* W_ih = (const float*)d_in[2];
    const float* W_hh = (const float*)d_in[3];
    const float* b_ih = (const float*)d_in[4];
    const float* b_hh = (const float*)d_in[5];
    const float* fc_W = (const float*)d_in[6];
    const float* fc_b = (const float*)d_in[7];
    float* out = (float*)d_out;
    unsigned short* ws = (unsigned short*)d_ws;

    hipLaunchKernelGGL(pack_weights_kernel, dim3(N_FRAG_TOT), dim3(32), 0, stream,
                       W_ih, W_hh, fc_W, ws);
    hipLaunchKernelGGL(gru_rollout_kernel, dim3(B_SZ / 16), dim3(512), 0, stream,
                       z, b_ih, b_hh, fc_b, ws, out);
}
